// GCN_78477642432529
// MI455X (gfx1250) — matrix operand order repaired and checked by reading, not compiled, not run on hardware
//
#include <hip/hip_runtime.h>

// ---------------------------------------------------------------------------
// GCN forward on MI455X (gfx1250), bf16 WMMA pipeline with LDS-staged GEMMs.
//
// o = relu(A @ (relu(A @ (point@W1^T + b1)) @ W2^T + b2)),  A = D^-1/2 E D^-1/2
//
// ~185 GFLOP total; compute-bound at dense-bf16 WMMA (~40us) vs ~12us/pass
// HBM for edge. Normalized adjacency pre-converted to bf16 (134MB) + hT
// (16MB) fit in the 192MB L2, so the big GEMMs stream from L2. LDS
// block-tile staging (128x256 per 8-wave block, double buffered) raises
// arithmetic intensity vs L2 to ~89 FLOP/B. Staging uses the CDNA5 async
// global->LDS DMA path (GLOBAL_LOAD_ASYNC_TO_LDS_B128, ASYNCcnt) when the
// toolchain exposes it, falling back to global->VGPR->LDS otherwise.
// ---------------------------------------------------------------------------

typedef __attribute__((ext_vector_type(16))) __bf16 v16bf;
typedef __attribute__((ext_vector_type(8)))  __bf16 v8bf;
typedef __attribute__((ext_vector_type(8)))  float  v8f;
typedef __attribute__((ext_vector_type(4)))  unsigned int v4u;
typedef __attribute__((ext_vector_type(4)))  int v4i;

#if defined(__gfx1250__) &&                                        \
    __has_builtin(__builtin_amdgcn_global_load_async_to_lds_b128) && \
    __has_builtin(__builtin_amdgcn_s_wait_asynccnt)
#define GCN_ASYNC_LDS 1
typedef __attribute__((address_space(1))) v4i as1_v4i;  // global int4
typedef __attribute__((address_space(3))) v4i as3_v4i;  // LDS int4
#else
#define GCN_ASYNC_LDS 0
#endif

#define N_NODES 8192
#define IN_C    512
#define HID_C   1024
#define OUT_C   256

__device__ __forceinline__ unsigned short f32_to_bf16_rne(float f) {
  union { float f; unsigned u; } x; x.f = f;
  unsigned u = x.u;
  return (unsigned short)((u + 0x7FFFu + ((u >> 16) & 1u)) >> 16);
}

// ---------------------------------------------------------------------------
// Kernel 1: row degree -> inv_sqrt.  One block per row of edge[8192][8192].
// ---------------------------------------------------------------------------
__global__ __launch_bounds__(256) void gcn_deg_kernel(
    const float* __restrict__ edge, float* __restrict__ inv_sqrt) {
  __shared__ float red[256];
  const int row = blockIdx.x;
  const float* p = edge + (size_t)row * N_NODES;
  float s = 0.f;
  for (int q = threadIdx.x; q < N_NODES / 4; q += 256) {
    float4 v = *(const float4*)(p + q * 4);
    s += v.x + v.y + v.z + v.w;
  }
  red[threadIdx.x] = s;
  __syncthreads();
  for (int off = 128; off > 0; off >>= 1) {
    if (threadIdx.x < off) red[threadIdx.x] += red[threadIdx.x + off];
    __syncthreads();
  }
  if (threadIdx.x == 0) {
    float d = red[0];
    inv_sqrt[row] = (d > 0.f) ? rsqrtf(d) : 0.f;
  }
}

// ---------------------------------------------------------------------------
// Kernel 2: A_bf16[i][j] = bf16( isq[i] * edge[i][j] * isq[j] )
// ---------------------------------------------------------------------------
__global__ __launch_bounds__(256) void gcn_normalize_kernel(
    const float* __restrict__ edge, const float* __restrict__ isq,
    unsigned short* __restrict__ A) {
  size_t q = (size_t)blockIdx.x * 256 + threadIdx.x;  // quad index
  size_t base = q * 4;
  int i = (int)(base >> 13);
  int j = (int)(base & (N_NODES - 1));
  float si = isq[i];
  float4 e = *(const float4*)(edge + base);
  unsigned long long r0 = f32_to_bf16_rne(si * isq[j + 0] * e.x);
  unsigned long long r1 = f32_to_bf16_rne(si * isq[j + 1] * e.y);
  unsigned long long r2 = f32_to_bf16_rne(si * isq[j + 2] * e.z);
  unsigned long long r3 = f32_to_bf16_rne(si * isq[j + 3] * e.w);
  *(unsigned long long*)(A + base) = r0 | (r1 << 16) | (r2 << 32) | (r3 << 48);
}

// ---------------------------------------------------------------------------
// Kernel 3: generic f32 -> bf16 conversion (n divisible by 4)
// ---------------------------------------------------------------------------
__global__ __launch_bounds__(256) void gcn_cvt_bf16_kernel(
    const float* __restrict__ in, unsigned short* __restrict__ out, int n) {
  int q = blockIdx.x * 256 + threadIdx.x;
  int base = q * 4;
  if (base >= n) return;
  float4 e = *(const float4*)(in + base);
  unsigned long long r0 = f32_to_bf16_rne(e.x);
  unsigned long long r1 = f32_to_bf16_rne(e.y);
  unsigned long long r2 = f32_to_bf16_rne(e.z);
  unsigned long long r3 = f32_to_bf16_rne(e.w);
  *(unsigned long long*)(out + base) = r0 | (r1 << 16) | (r2 << 32) | (r3 << 48);
}

// ---------------------------------------------------------------------------
// LDS-staged WMMA GEMM:  D[M][N] = epilogue( X[M][K] * Y[N][K]^T )
//   X, Y bf16 row-major (K contiguous). Block = WROWS x WCOLS wave32's.
//   Block tile BM x BN; wave tile (TM*16) x (TN*16); K-step KS=32.
//   Double-buffered LDS staging; padded rows (stride KS+8 halves = 80B)
//   for conflict-free ds_load_b128 fragments.
//   EPI: 0 = +bias[row], store bf16 | 1 = relu, store bf16 | 2 = relu, f32.
//
// Fragment layouts per CDNA5 ISA (CDNA5 ISA, wave32):
//   A 16x32 bf16: lane L: row M=L%16, K = {0..7,16..23} (+8 if L>=16).
//   B 32x16 bf16: lane L: col N=L%16, K = {0..7,16..23} (+8 if L>=16).
//   C/D 16x16 f32: VGPR r, lane L -> (M = r + (L>=16 ? 8:0), N = L%16).
// ---------------------------------------------------------------------------
template <int BM, int BN, int WROWS, int WCOLS, int TM, int TN, int EPI>
__global__ __launch_bounds__(WROWS * WCOLS * 32) void gcn_wmma_gemm_xyt(
    const unsigned short* __restrict__ X, const unsigned short* __restrict__ Y,
    const float* __restrict__ bias, void* __restrict__ D,
    int M, int N, int K) {
  constexpr int THREADS = WROWS * WCOLS * 32;
  constexpr int KS  = 32;          // K elements per staged tile
  constexpr int LDA = KS + 8;      // padded LDS row stride (halves)
  constexpr int KCH = KS / 8;      // 16B chunks per row
  constexpr int APT = BM * KCH / THREADS;  // A staging chunks per thread
  constexpr int BPT = BN * KCH / THREADS;  // B staging chunks per thread
  static_assert(BM == WROWS * TM * 16 && BN == WCOLS * TN * 16, "tile");
  static_assert((BM * KCH) % THREADS == 0 && (BN * KCH) % THREADS == 0, "chk");

  __shared__ __align__(16) unsigned short As[2][BM * LDA];
  __shared__ __align__(16) unsigned short Bs[2][BN * LDA];

  const int tid  = threadIdx.x;
  const int lane = tid & 31;
  const int w    = tid >> 5;
  const int wrow = w / WCOLS, wcol = w % WCOLS;
  const int mloc = wrow * TM * 16;
  const int nloc = wcol * TN * 16;
  const int lr   = lane & 15;
  const int hi   = lane >> 4;
  const int aoff = hi * 8;     // A-frag K sub-offset
  const int boff = hi * 8;     // B-frag K sub-offset
  const size_t bm0 = (size_t)blockIdx.y * BM;
  const size_t bn0 = (size_t)blockIdx.x * BN;

  // Staging maps: linear chunk -> (row, 8-elem col) of the block tile.
  const unsigned short* agp[APT]; int alds[APT];
  const unsigned short* bgp[BPT]; int blds[BPT];
#pragma unroll
  for (int t = 0; t < APT; t++) {
    int ch = tid + t * THREADS;
    int r = ch / KCH, c = (ch % KCH) * 8;
    agp[t]  = X + (bm0 + r) * (size_t)K + c;
    alds[t] = r * LDA + c;
  }
#pragma unroll
  for (int t = 0; t < BPT; t++) {
    int ch = tid + t * THREADS;
    int r = ch / KCH, c = (ch % KCH) * 8;
    bgp[t]  = Y + (bn0 + r) * (size_t)K + c;
    blds[t] = r * LDA + c;
  }

  v8f acc[TM][TN];
  const v8f vzero = {0.f, 0.f, 0.f, 0.f, 0.f, 0.f, 0.f, 0.f};
#pragma unroll
  for (int i = 0; i < TM; i++)
#pragma unroll
    for (int j = 0; j < TN; j++) acc[i][j] = vzero;

  // Prologue: stage tile 0 into buffer 0.
#if GCN_ASYNC_LDS
#pragma unroll
  for (int t = 0; t < APT; t++)
    __builtin_amdgcn_global_load_async_to_lds_b128(
        (as1_v4i*)(void*)agp[t], (as3_v4i*)(void*)&As[0][alds[t]], 0, 0);
#pragma unroll
  for (int t = 0; t < BPT; t++)
    __builtin_amdgcn_global_load_async_to_lds_b128(
        (as1_v4i*)(void*)bgp[t], (as3_v4i*)(void*)&Bs[0][blds[t]], 0, 0);
  __builtin_amdgcn_s_wait_asynccnt(0);
  __syncthreads();
#else
  v4u ar[APT], br[BPT];
#pragma unroll
  for (int t = 0; t < APT; t++) ar[t] = *(const v4u*)(const void*)agp[t];
#pragma unroll
  for (int t = 0; t < BPT; t++) br[t] = *(const v4u*)(const void*)bgp[t];
#pragma unroll
  for (int t = 0; t < APT; t++) *(v4u*)(void*)&As[0][alds[t]] = ar[t];
#pragma unroll
  for (int t = 0; t < BPT; t++) *(v4u*)(void*)&Bs[0][blds[t]] = br[t];
  __syncthreads();
#endif

  const int nK = K / KS;
  for (int kt = 0; kt < nK; ++kt) {
    const int cur  = kt & 1;
    const bool more = (kt + 1) < nK;
    if (more) {
      const int kn = (kt + 1) * KS;
#if GCN_ASYNC_LDS
      // Issue async DMA for the next tile now; buf[next] readers finished
      // before the previous barrier, so writing it here is race-free.
      const int nxt = 1 - cur;
#pragma unroll
      for (int t = 0; t < APT; t++) {
        __builtin_prefetch((const void*)(agp[t] + kn + KS), 0, 1);
        __builtin_amdgcn_global_load_async_to_lds_b128(
            (as1_v4i*)(void*)(agp[t] + kn),
            (as3_v4i*)(void*)&As[nxt][alds[t]], 0, 0);
      }
#pragma unroll
      for (int t = 0; t < BPT; t++) {
        __builtin_prefetch((const void*)(bgp[t] + kn + KS), 0, 1);
        __builtin_amdgcn_global_load_async_to_lds_b128(
            (as1_v4i*)(void*)(bgp[t] + kn),
            (as3_v4i*)(void*)&Bs[nxt][blds[t]], 0, 0);
      }
#else
#pragma unroll
      for (int t = 0; t < APT; t++) {
        __builtin_prefetch((const void*)(agp[t] + kn + KS), 0, 1);
        ar[t] = *(const v4u*)(const void*)(agp[t] + kn);
      }
#pragma unroll
      for (int t = 0; t < BPT; t++) {
        __builtin_prefetch((const void*)(bgp[t] + kn + KS), 0, 1);
        br[t] = *(const v4u*)(const void*)(bgp[t] + kn);
      }
#endif
    }

    // Fragments from LDS + WMMA.
    v16bf af[TM], bfr[TN];
#pragma unroll
    for (int t = 0; t < TM; t++) {
      const unsigned short* p = &As[cur][(mloc + t * 16 + lr) * LDA + aoff];
      union { v16bf v; v8bf h[2]; } u;
      u.h[0] = *(const v8bf*)(const void*)(p);
      u.h[1] = *(const v8bf*)(const void*)(p + 16);
      af[t] = u.v;
    }
#pragma unroll
    for (int t = 0; t < TN; t++) {
      const unsigned short* p = &Bs[cur][(nloc + t * 16 + lr) * LDA + boff];
      union { v16bf v; v8bf h[2]; } u;
      u.h[0] = *(const v8bf*)(const void*)(p);
      u.h[1] = *(const v8bf*)(const void*)(p + 16);
      bfr[t] = u.v;
    }
#pragma unroll
    for (int i = 0; i < TM; i++)
#pragma unroll
      for (int j = 0; j < TN; j++)
        acc[i][j] = __builtin_amdgcn_wmma_f32_16x16x32_bf16(
            false, af[i], false, bfr[j], (short)0, acc[i][j], false, false);

    if (more) {
#if GCN_ASYNC_LDS
      __builtin_amdgcn_s_wait_asynccnt(0);  // my DMA into buf[next] done
      __syncthreads();                      // everyone's DMA done + cur freed
#else
      __syncthreads();  // all waves done reading buf[cur]
      const int nxt = 1 - cur;
#pragma unroll
      for (int t = 0; t < APT; t++) *(v4u*)(void*)&As[nxt][alds[t]] = ar[t];
#pragma unroll
      for (int t = 0; t < BPT; t++) *(v4u*)(void*)&Bs[nxt][blds[t]] = br[t];
      __syncthreads();
#endif
    }
  }

  // Epilogue + store
#pragma unroll
  for (int i = 0; i < TM; i++) {
    const int mbase = (int)bm0 + mloc + i * 16 + hi * 8;
#pragma unroll
    for (int j = 0; j < TN; j++) {
      const int col = (int)bn0 + nloc + j * 16 + lr;
#pragma unroll
      for (int r = 0; r < 8; r++) {
        const int row = mbase + r;
        float v = acc[i][j][r];
        if (EPI == 0) v += bias[row];
        else          v = fmaxf(v, 0.f);
        if (EPI == 2)
          ((float*)D)[(size_t)row * N + col] = v;
        else
          ((unsigned short*)D)[(size_t)row * N + col] = f32_to_bf16_rne(v);
      }
    }
  }
}

// ---------------------------------------------------------------------------
// Launch
// ---------------------------------------------------------------------------
extern "C" void kernel_launch(void* const* d_in, const int* in_sizes, int n_in,
                              void* d_out, int out_size, void* d_ws,
                              size_t ws_size, hipStream_t stream) {
  (void)in_sizes; (void)n_in; (void)out_size; (void)ws_size;

  const float* point = (const float*)d_in[0];  // [8192, 512]
  const float* edge  = (const float*)d_in[1];  // [8192, 8192]
  const float* W1    = (const float*)d_in[2];  // [1024, 512]
  const float* b1    = (const float*)d_in[3];  // [1024]
  const float* W2    = (const float*)d_in[4];  // [256, 1024]
  const float* b2    = (const float*)d_in[5];  // [256]
  float* out = (float*)d_out;                  // [8192, 256]

  char* ws = (char*)d_ws;
  size_t off = 0;
  float* isq = (float*)(ws + off);                    off += (size_t)N_NODES * 4;
  unsigned short* Abf  = (unsigned short*)(ws + off); off += (size_t)N_NODES * N_NODES * 2;
  unsigned short* ptbf = (unsigned short*)(ws + off); off += (size_t)N_NODES * IN_C * 2;
  unsigned short* w1bf = (unsigned short*)(ws + off); off += (size_t)HID_C * IN_C * 2;
  unsigned short* w2bf = (unsigned short*)(ws + off); off += (size_t)OUT_C * HID_C * 2;
  unsigned short* hT   = (unsigned short*)(ws + off); off += (size_t)HID_C * N_NODES * 2;
  unsigned short* h2   = (unsigned short*)(ws + off); off += (size_t)N_NODES * HID_C * 2;
  unsigned short* oT   = (unsigned short*)(ws + off); off += (size_t)OUT_C * N_NODES * 2;

  // 1) degrees -> D^-1/2
  gcn_deg_kernel<<<N_NODES, 256, 0, stream>>>(edge, isq);

  // 2) A_bf16 = bf16(isq_i * edge * isq_j)
  gcn_normalize_kernel<<<(N_NODES * (size_t)N_NODES) / 4 / 256, 256, 0, stream>>>(
      edge, isq, Abf);

  // 3) f32 -> bf16 conversions
  gcn_cvt_bf16_kernel<<<(N_NODES * IN_C) / 4 / 256, 256, 0, stream>>>(
      point, ptbf, N_NODES * IN_C);
  gcn_cvt_bf16_kernel<<<(HID_C * IN_C) / 4 / 256, 256, 0, stream>>>(
      W1, w1bf, HID_C * IN_C);
  gcn_cvt_bf16_kernel<<<(OUT_C * HID_C) / 4 / 256, 256, 0, stream>>>(
      W2, w2bf, OUT_C * HID_C);

  // 4) hT[c][j] = point[j]·W1[c] + b1[c]     M=1024 N=8192 K=512
  gcn_wmma_gemm_xyt<128, 256, 2, 4, 4, 4, 0>
      <<<dim3(N_NODES / 256, HID_C / 128), 256, 0, stream>>>(
          w1bf, ptbf, b1, hT, HID_C, N_NODES, IN_C);

  // 5) h2[i][c] = relu( A[i]·hT[c] )         M=8192 N=1024 K=8192
  gcn_wmma_gemm_xyt<128, 256, 2, 4, 4, 4, 1>
      <<<dim3(HID_C / 256, N_NODES / 128), 256, 0, stream>>>(
          Abf, hT, nullptr, h2, N_NODES, HID_C, N_NODES);

  // 6) oT[c2][j] = h2[j]·W2[c2] + b2[c2]     M=256 N=8192 K=1024
  gcn_wmma_gemm_xyt<128, 256, 2, 4, 4, 4, 0>
      <<<dim3(N_NODES / 256, OUT_C / 128), 256, 0, stream>>>(
          w2bf, h2, b2, oT, OUT_C, N_NODES, HID_C);

  // 7) out[i][c2] = relu( A[i]·oT[c2] )      M=8192 N=256 K=8192 (f32 out)
  gcn_wmma_gemm_xyt<128, 64, 4, 2, 2, 2, 2>
      <<<dim3(OUT_C / 64, N_NODES / 128), 256, 0, stream>>>(
          Abf, oT, nullptr, out, N_NODES, OUT_C, N_NODES);
}
